// MDRMWithCPRecon_60644938220018
// MI455X (gfx1250) — matrix operand order repaired and checked by reading, not compiled, not run on hardware
//
#include <hip/hip_runtime.h>
#include <hip/hip_bf16.h>

typedef __attribute__((ext_vector_type(16))) _Float16 v16h;
typedef __attribute__((ext_vector_type(8)))  _Float16 v8h;
typedef __attribute__((ext_vector_type(8)))  float    v8f;

#define BB 8
#define CC 256
#define HH 128
#define WW 128
#define KK 4
#define CIN 512   // 2*C

// ---------------- workspace layout (bytes) ----------------
constexpr size_t SZ_XH = (size_t)BB*HH*WW*CIN*2;      // 134217728  f16 NHWC input concat
constexpr size_t SZ_WT = (size_t)9*CC*CIN*2;          // 2359296    f16 [tap][co][ci]
constexpr size_t SZ_FM = (size_t)BB*HH*WW*CC*2;       // 67108864   f16 NHWC Fm
constexpr size_t SZ_PS = (size_t)BB*64*CC*4;          // 524288     mode1 partials
constexpr size_t OFF_XH = 0;
constexpr size_t OFF_WT = OFF_XH + SZ_XH;
constexpr size_t OFF_FM = OFF_WT + SZ_WT;
constexpr size_t OFF_PS = OFF_FM + SZ_FM;
constexpr size_t OFF_PM = OFF_PS + SZ_PS;
constexpr size_t OFF_U1 = OFF_PM + SZ_PS;             // B*C*K f32
constexpr size_t OFF_U2 = OFF_U1 + (size_t)BB*CC*KK*4;
constexpr size_t OFF_U3 = OFF_U2 + (size_t)BB*WW*KK*4;
constexpr size_t OFF_CV = OFF_U3 + (size_t)BB*HH*KK*4; // 36 floats -> pad 256
constexpr size_t OFF_GA = OFF_CV + 256;
constexpr size_t OFF_GM = OFF_GA + (size_t)BB*CC*4;
constexpr size_t OFF_SP = OFF_GM + (size_t)BB*CC*4;
constexpr size_t OFF_M1 = OFF_SP + (size_t)BB*CC*4;

// ---------------- pack kernels ----------------
#define PPAD 258   // LDS tile pitch in halfs: PPAD/2 = 129 == 1 mod 64 banks
__global__ __launch_bounds__(256) void k_pack_x(const float* __restrict__ frm,
                                                const float* __restrict__ oth,
                                                _Float16* __restrict__ xh) {
  // one block per (b, 64-pixel group); LDS transpose so both sides are coalesced
  __shared__ __align__(16) _Float16 tile[64 * PPAD];
  int blk = blockIdx.x;
  int b  = blk >> 8;
  int p0 = (blk & 255) * 64;
  int t  = threadIdx.x;
  int px = t & 63;
  int cq = t >> 6;            // 0..3
  int q2 = t >> 7;            // 0..1
  int c2 = (t & 127) * 2;
  for (int half = 0; half < 2; ++half) {
    __syncthreads();
    const float* planebase = (half == 0) ? frm : oth;
    for (int i = 0; i < 64; ++i) {
      int cl = i*4 + cq;
      const float* src = planebase + ((size_t)(b*CC + cl))*(HH*WW);
      tile[px*PPAD + cl] = (_Float16)src[p0 + px];
    }
    __syncthreads();
    for (int j = 0; j < 32; ++j) {
      int q = j*2 + q2;
      unsigned v = *reinterpret_cast<const unsigned*>(&tile[q*PPAD + c2]);
      *reinterpret_cast<unsigned*>(&xh[((size_t)(b*(HH*WW) + p0 + q))*CIN + half*CC + c2]) = v;
    }
  }
}

__global__ __launch_bounds__(256) void k_pack_w(const float* __restrict__ cw,
                                                _Float16* __restrict__ wt) {
  size_t i = (size_t)blockIdx.x*256 + threadIdx.x;   // 9*256*512 total
  if (i >= (size_t)9*CC*CIN) return;
  int ci = i & (CIN-1);
  int rest = (int)(i >> 9);
  int co = rest & (CC-1);
  int t  = rest >> 8;
  // conv3x3_w is [co][ci][ky][kx]
  wt[i] = (_Float16)cw[((size_t)co*CIN + ci)*9 + t];
}

// collapse adapters+ug into 36 scalars: cvec[(adapter*3 + {wa,wm,wb})*4 + k]
__global__ void k_adapters(const float* a1w, const float* a1b, const float* a2w, const float* a2b,
                           const float* a3w, const float* a3b, const float* ugw, const float* ugb,
                           float* cvec) {
  int t = threadIdx.x;
  if (t >= 36) return;
  int i = t / 12, r = t % 12, j = r / 4, k = r % 4;
  const float* aw = (i == 0) ? a1w : (i == 1) ? a2w : a3w;
  const float* ab = (i == 0) ? a1b : (i == 1) ? a2b : a3b;
  float s = 0.f;
  for (int c = 0; c < CC; ++c) {
    float u = ugw[k*CC + c];
    float v = (j == 0) ? aw[c*2 + 0] : (j == 1) ? aw[c*2 + 1] : ab[c];
    s += u * v;
  }
  if (j == 2) s += ugb[k];
  cvec[(i*3 + j)*4 + k] = s;
}

// ---------------- WMMA implicit-GEMM conv3x3 + bias + LeakyReLU ----------------
// Double-buffered LDS, staged with CDNA5 async global->LDS DMA (ASYNCcnt).
__global__ __launch_bounds__(256) void conv_wmma(const _Float16* __restrict__ xh,
                                                 const _Float16* __restrict__ wt,
                                                 const float* __restrict__ bias,
                                                 _Float16* __restrict__ Fm) {
  __shared__ __align__(16) _Float16 As[2][64*32];    // [buf][pixel m][k]
  __shared__ __align__(16) _Float16 Bs[2][128*32];   // [buf][cout n][k] (transposed)
  const int tid  = threadIdx.x;
  const int lane = tid & 31;
  const int wv   = tid >> 5;
  const int wm   = wv & 1;            // M half (2 x 32)
  const int wn   = wv >> 1;           // N quarter (4 x 32)
  const int gx   = blockIdx.x;        // 0..3 : (wtile, ntile)
  const int w0   = (gx & 1) * 64;
  const int co0  = (gx >> 1) * 128;
  const int h    = blockIdx.y;
  const int b    = blockIdx.z;

  v8f acc[2][2] = {};

  const int ar = tid >> 2;            // A stage row 0..63
  const int ac = (tid & 3) * 8;       // A stage col chunk (8 halfs = 16B)
  const int br = tid >> 1;            // B stage row 0..127
  const int bc = (tid & 1) * 16;      // B stage col chunk (16 halfs = 32B)

  // wave-relative LDS byte offsets for async DMA destinations
  unsigned ldsA[2], ldsB[2];
  ldsA[0] = (unsigned)(size_t)&As[0][ar*32 + ac];
  ldsA[1] = (unsigned)(size_t)&As[1][ar*32 + ac];
  ldsB[0] = (unsigned)(size_t)&Bs[0][br*32 + bc];
  ldsB[1] = (unsigned)(size_t)&Bs[1][br*32 + bc];

  auto issue = [&](int ks, int buf) {
    const int t   = ks >> 4;
    const int ci0 = (ks & 15) << 5;
    const int ky  = t / 3, kx = t - ky*3;
    const int y   = h + ky - 1;
    const int xA  = w0 + ar + kx - 1;
    if (((unsigned)y < HH) && ((unsigned)xA < WW)) {
      unsigned long long gA = (unsigned long long)(size_t)
          (xh + (((size_t)(b*HH + y)*WW + xA) << 9) + ci0 + ac);
      asm volatile("global_load_async_to_lds_b128 %0, %1, off"
                   :: "v"(ldsA[buf]), "v"(gA) : "memory");
    } else {
      *reinterpret_cast<uint4*>(&As[buf][ar*32 + ac]) = make_uint4(0u, 0u, 0u, 0u);
    }
    unsigned long long gB = (unsigned long long)(size_t)
        (wt + ((size_t)(t*CC + co0 + br) << 9) + ci0 + bc);
    // INST_OFFSET is added to both LDS and memory address (ISA 10.x async addressing)
    asm volatile("global_load_async_to_lds_b128 %0, %1, off"
                 :: "v"(ldsB[buf]), "v"(gB) : "memory");
    asm volatile("global_load_async_to_lds_b128 %0, %1, off offset:16"
                 :: "v"(ldsB[buf]), "v"(gB) : "memory");
  };

  const int mrow = lane & 15;
  const int ka   = (lane >= 16) ? 16 : 0;   // A lane-half K offset (contiguous k = 16h + s, matches B)
  const int kb   = (lane >= 16) ? 16 : 0;   // B lane-half K offset (contiguous k = 16h + s, matches A)

  issue(0, 0);
  asm volatile("s_wait_asynccnt 0" ::: "memory");
  __syncthreads();

  for (int ks = 0; ks < 144; ++ks) {
    const int cur = ks & 1;
    if (ks + 1 < 144) issue(ks + 1, cur ^ 1);   // prefetch next k-step into other buffer

    v16h afr[2], bfr[2];
#pragma unroll
    for (int i = 0; i < 2; ++i) {
      const int r = wm*32 + i*16 + mrow;
      v8h lo = *reinterpret_cast<const v8h*>(&As[cur][r*32 + ka]);
      v8h hi = *reinterpret_cast<const v8h*>(&As[cur][r*32 + ka + 8]);
      afr[i] = __builtin_shufflevector(lo, hi, 0,1,2,3,4,5,6,7,8,9,10,11,12,13,14,15);
    }
#pragma unroll
    for (int j = 0; j < 2; ++j) {
      const int n = wn*32 + j*16 + mrow;
      v8h lo = *reinterpret_cast<const v8h*>(&Bs[cur][n*32 + kb]);
      v8h hi = *reinterpret_cast<const v8h*>(&Bs[cur][n*32 + kb + 8]);
      bfr[j] = __builtin_shufflevector(lo, hi, 0,1,2,3,4,5,6,7,8,9,10,11,12,13,14,15);
    }
#pragma unroll
    for (int i = 0; i < 2; ++i)
#pragma unroll
      for (int j = 0; j < 2; ++j)
        acc[i][j] = __builtin_amdgcn_wmma_f32_16x16x32_f16(
            false, afr[i], false, bfr[j], (short)0, acc[i][j], false, false);

    asm volatile("s_wait_asynccnt 0" ::: "memory");  // next buffer staged
    __syncthreads();
  }

  // epilogue: bias + LeakyReLU(0.01), store NHWC f16 (cout coalesced)
  const int nlo = lane & 15;
  const int mo  = (lane >= 16) ? 8 : 0;
#pragma unroll
  for (int i = 0; i < 2; ++i)
#pragma unroll
    for (int j = 0; j < 2; ++j) {
      const int n  = co0 + wn*32 + j*16 + nlo;
      const float bn = bias[n];
#pragma unroll
      for (int v = 0; v < 8; ++v) {
        const int m = wm*32 + i*16 + v + mo;
        float val = acc[i][j][v] + bn;
        val = (val >= 0.f) ? val : 0.01f * val;
        Fm[(((size_t)(b*HH + h)*WW) + w0 + m)*CC + n] = (_Float16)val;
      }
    }
}

// ---------------- mode reductions + U softmax ----------------
__global__ __launch_bounds__(256) void k_mode1_part(const _Float16* __restrict__ Fm,
                                                    float* __restrict__ ps, float* __restrict__ pm) {
  int blk = blockIdx.x, b = blk >> 6, ch = blk & 63;  // 64 pixel-chunks of 256
  int c = threadIdx.x;
  const _Float16* base = Fm + ((size_t)(b*(HH*WW) + ch*256))*CC + c;
  float s = 0.f, m = -3.4e38f;
  for (int p = 0; p < 256; ++p) {
    float v = (float)base[(size_t)p*CC];
    s += v; m = fmaxf(m, v);
  }
  ps[(size_t)(b*64 + ch)*CC + c] = s;
  pm[(size_t)(b*64 + ch)*CC + c] = m;
}

__device__ __forceinline__ void softmax4_store(float avg, float mx, const float* cv, float* dst) {
  float lg[4], lmax = -3.4e38f;
#pragma unroll
  for (int k = 0; k < 4; ++k) { lg[k] = avg*cv[k] + mx*cv[4+k] + cv[8+k]; lmax = fmaxf(lmax, lg[k]); }
  float e[4], den = 0.f;
#pragma unroll
  for (int k = 0; k < 4; ++k) { e[k] = expf(lg[k] - lmax); den += e[k]; }
  float inv = 1.f / den;
#pragma unroll
  for (int k = 0; k < 4; ++k) dst[k] = e[k] * inv;
}

__global__ void k_U1(const float* __restrict__ ps, const float* __restrict__ pm,
                     const float* __restrict__ cvec, float* __restrict__ U1) {
  int b = blockIdx.x, c = threadIdx.x;
  float s = 0.f, m = -3.4e38f;
  for (int ch = 0; ch < 64; ++ch) {
    s += ps[(size_t)(b*64 + ch)*CC + c];
    m  = fmaxf(m, pm[(size_t)(b*64 + ch)*CC + c]);
  }
  softmax4_store(s * (1.f/(HH*WW)), m, cvec + 0, &U1[(size_t)(b*CC + c)*4]);
}

__global__ __launch_bounds__(256) void k_U2(const _Float16* __restrict__ Fm,
                                            const float* __restrict__ cvec, float* __restrict__ U2) {
  __shared__ float ssum[256], smax[256];
  int blk = blockIdx.x, b = blk >> 7, w = blk & 127;
  int c = threadIdx.x;
  float s = 0.f, m = -3.4e38f;
  for (int h = 0; h < HH; ++h) {
    float v = (float)Fm[((size_t)((b*HH + h)*WW) + w)*CC + c];
    s += v; m = fmaxf(m, v);
  }
  ssum[c] = s; smax[c] = m; __syncthreads();
  for (int st = 128; st > 0; st >>= 1) {
    if (c < st) { ssum[c] += ssum[c+st]; smax[c] = fmaxf(smax[c], smax[c+st]); }
    __syncthreads();
  }
  if (c == 0)
    softmax4_store(ssum[0] * (1.f/(CC*HH)), smax[0], cvec + 12, &U2[(size_t)(b*WW + w)*4]);
}

__global__ __launch_bounds__(256) void k_U3(const _Float16* __restrict__ Fm,
                                            const float* __restrict__ cvec, float* __restrict__ U3) {
  __shared__ float ssum[256], smax[256];
  int blk = blockIdx.x, b = blk >> 7, h = blk & 127;
  int c = threadIdx.x;
  float s = 0.f, m = -3.4e38f;
  for (int w = 0; w < WW; ++w) {
    float v = (float)Fm[((size_t)((b*HH + h)*WW) + w)*CC + c];
    s += v; m = fmaxf(m, v);
  }
  ssum[c] = s; smax[c] = m; __syncthreads();
  for (int st = 128; st > 0; st >>= 1) {
    if (c < st) { ssum[c] += ssum[c+st]; smax[c] = fmaxf(smax[c], smax[c+st]); }
    __syncthreads();
  }
  if (c == 0)
    softmax4_store(ssum[0] * (1.f/(CC*WW)), smax[0], cvec + 24, &U3[(size_t)(b*HH + h)*4]);
}

// ---------------- spectral path ----------------
__global__ void k_gstats(const float* __restrict__ U1, const float* __restrict__ U2,
                         const float* __restrict__ U3, float* __restrict__ gavg, float* __restrict__ gmax) {
  int b = blockIdx.x, c = threadIdx.x;
  const float* u1 = &U1[(size_t)(b*CC + c)*4];
  float u0 = u1[0], u1v = u1[1], u2v = u1[2], u3v = u1[3];
  float s = 0.f, m = -3.4e38f;
  for (int n = 0; n < WW + HH; ++n) {
    const float* un = (n < WW) ? &U2[(size_t)(b*WW + n)*4] : &U3[(size_t)(b*HH + (n - WW))*4];
    float v = u0*un[0] + u1v*un[1] + u2v*un[2] + u3v*un[3];
    s += v; m = fmaxf(m, v);
  }
  gavg[b*CC + c] = s * (1.f/(WW + HH));
  gmax[b*CC + c] = m;
}

__global__ void k_spec(const float* __restrict__ gavg, const float* __restrict__ gmax,
                       const float* __restrict__ spaw, const float* __restrict__ spab,
                       const float* __restrict__ spmw, const float* __restrict__ spmb,
                       float* __restrict__ spec) {
  int b = blockIdx.x, o = threadIdx.x;
  float acc = spab[o] + spmb[o];
  for (int c = 0; c < CC; ++c)
    acc += spaw[(size_t)o*CC + c]*gavg[b*CC + c] + spmw[(size_t)o*CC + c]*gmax[b*CC + c];
  float s = 1.f / (1.f + expf(-acc));
  spec[b*CC + o] = 1.f / (1.f + expf(-s));     // double sigmoid as in reference
}

__global__ void k_M1(const float* __restrict__ reconw, const float* __restrict__ U1,
                     float* __restrict__ M1) {
  int b = blockIdx.x, o = threadIdx.x;
  float m0 = 0.f, m1 = 0.f, m2 = 0.f, m3 = 0.f;
  for (int c = 0; c < CC; ++c) {
    float w = reconw[(size_t)o*CC + c];
    const float* u = &U1[(size_t)(b*CC + c)*4];
    m0 += w*u[0]; m1 += w*u[1]; m2 += w*u[2]; m3 += w*u[3];
  }
  float* d = &M1[(size_t)(b*CC + o)*4];
  d[0] = m0; d[1] = m1; d[2] = m2; d[3] = m3;
}

// ---------------- fused output: fused + cp (both NCHW f32) ----------------
__global__ __launch_bounds__(256) void k_fuse(const float* __restrict__ frm, const float* __restrict__ oth,
                                              const float* __restrict__ U2, const float* __restrict__ U3,
                                              const float* __restrict__ spec, const float* __restrict__ M1,
                                              const float* __restrict__ reconb,
                                              const float* __restrict__ spw, const float* __restrict__ spb,
                                              const float* __restrict__ alphap,
                                              float* __restrict__ out_fused, float* __restrict__ out_cp) {
  int blk = blockIdx.x, b = blk >> 7, h = blk & 127;
  int t = threadIdx.x, w = t & 127, coff = t >> 7;
  const float* u2 = &U2[(size_t)(b*WW + w)*4];
  const float* u3 = &U3[(size_t)(b*HH + h)*4];
  float s0 = u2[0]*u3[0], s1 = u2[1]*u3[1], s2 = u2[2]*u3[2], s3 = u2[3]*u3[3];
  float dot = s0 + s1 + s2 + s3;
  float spat = 1.f / (1.f + expf(-(spw[0]*dot + spb[0])));
  float alpha = alphap[0];
  for (int c = coff; c < CC; c += 2) {
    size_t idx = (((size_t)(b*CC + c)*HH) + h)*WW + w;
    float Watt = spec[b*CC + c] * spat;
    out_fused[idx] = alpha*Watt*frm[idx] + (1.f - alpha)*(1.f - Watt)*oth[idx];
    const float* mm = &M1[(size_t)(b*CC + c)*4];
    out_cp[idx] = mm[0]*s0 + mm[1]*s1 + mm[2]*s2 + mm[3]*s3 + reconb[c];
  }
}

// ---------------- host launcher ----------------
extern "C" void kernel_launch(void* const* d_in, const int* in_sizes, int n_in,
                              void* d_out, int out_size, void* d_ws, size_t ws_size,
                              hipStream_t stream) {
  (void)in_sizes; (void)n_in; (void)out_size; (void)ws_size;
  const float* frm    = (const float*)d_in[0];
  const float* oth    = (const float*)d_in[1];
  const float* convw  = (const float*)d_in[2];
  const float* convb  = (const float*)d_in[3];
  const float* a1w    = (const float*)d_in[4];
  const float* a1b    = (const float*)d_in[5];
  const float* a2w    = (const float*)d_in[6];
  const float* a2b    = (const float*)d_in[7];
  const float* a3w    = (const float*)d_in[8];
  const float* a3b    = (const float*)d_in[9];
  const float* ugw    = (const float*)d_in[10];
  const float* ugb    = (const float*)d_in[11];
  const float* reconw = (const float*)d_in[12];
  const float* reconb = (const float*)d_in[13];
  const float* spw    = (const float*)d_in[14];
  const float* spb    = (const float*)d_in[15];
  const float* spaw   = (const float*)d_in[16];
  const float* spab   = (const float*)d_in[17];
  const float* spmw   = (const float*)d_in[18];
  const float* spmb   = (const float*)d_in[19];
  const float* alphap = (const float*)d_in[20];

  char* ws = (char*)d_ws;
  _Float16* xh  = (_Float16*)(ws + OFF_XH);
  _Float16* wtp = (_Float16*)(ws + OFF_WT);
  _Float16* Fm  = (_Float16*)(ws + OFF_FM);
  float* ps   = (float*)(ws + OFF_PS);
  float* pm   = (float*)(ws + OFF_PM);
  float* U1   = (float*)(ws + OFF_U1);
  float* U2   = (float*)(ws + OFF_U2);
  float* U3   = (float*)(ws + OFF_U3);
  float* cvec = (float*)(ws + OFF_CV);
  float* gavg = (float*)(ws + OFF_GA);
  float* gmax = (float*)(ws + OFF_GM);
  float* spec = (float*)(ws + OFF_SP);
  float* M1   = (float*)(ws + OFF_M1);

  float* out_fused = (float*)d_out;
  float* out_cp    = out_fused + (size_t)BB*CC*HH*WW;

  k_pack_x<<<dim3(BB*256), 256, 0, stream>>>(frm, oth, xh);
  k_pack_w<<<dim3((9*CC*CIN + 255)/256), 256, 0, stream>>>(convw, wtp);
  k_adapters<<<1, 64, 0, stream>>>(a1w, a1b, a2w, a2b, a3w, a3b, ugw, ugb, cvec);
  conv_wmma<<<dim3(4, HH, BB), 256, 0, stream>>>(xh, wtp, convb, Fm);
  k_mode1_part<<<dim3(BB*64), 256, 0, stream>>>(Fm, ps, pm);
  k_U1<<<dim3(BB), 256, 0, stream>>>(ps, pm, cvec, U1);
  k_U2<<<dim3(BB*WW), 256, 0, stream>>>(Fm, cvec, U2);
  k_U3<<<dim3(BB*HH), 256, 0, stream>>>(Fm, cvec, U3);
  k_gstats<<<dim3(BB), 256, 0, stream>>>(U1, U2, U3, gavg, gmax);
  k_spec<<<dim3(BB), 256, 0, stream>>>(gavg, gmax, spaw, spab, spmw, spmb, spec);
  k_M1<<<dim3(BB), 256, 0, stream>>>(reconw, U1, M1);
  k_fuse<<<dim3(BB*HH), 256, 0, stream>>>(frm, oth, U2, U3, spec, M1, reconb,
                                          spw, spb, alphap, out_fused, out_cp);
}
